// LowRankAttention_15994458211055
// MI455X (gfx1250) — matrix operand order repaired and checked by reading, not compiled, not run on hardware
//
#include <hip/hip_runtime.h>
#include <hip/hip_bf16.h>

// ---------------------------------------------------------------------------
// LowRankAttention on MI455X (gfx1250, wave32, WMMA 16x16x32 bf16, f32 acc)
// bf16x3 split-precision GEMMs (xh*wh + xl*wh + xh*wl) ~= fp32 accuracy.
// ---------------------------------------------------------------------------

typedef __attribute__((ext_vector_type(16))) __bf16 v16bf;
typedef __attribute__((ext_vector_type(8)))  float  v8f;
typedef __attribute__((ext_vector_type(4)))  float  v4f;
typedef __attribute__((ext_vector_type(4)))  unsigned int v4u;

#define LDK 40          // padded LDS row stride (bf16 elems) for a 32-wide K tile
#define NROWS 65536
#define DDIM 1024
#define KDIM 256

__device__ __forceinline__ void split_bf16(float x, __bf16& h, __bf16& l) {
    h = (__bf16)x;
    l = (__bf16)(x - (float)h);
}

__device__ __forceinline__ v8f wmma_bf16(v16bf a, v16bf b, v8f c) {
    // v_wmma_f32_16x16x32_bf16
    return __builtin_amdgcn_wmma_f32_16x16x32_bf16(
        false, a, false, b, (short)0, c, false, false);
}

// A-matrix 16x32 bf16 fragment: lanes 0-15 hold K 0..15 (V0-7);
// lanes 16-31 hold K 16..31. Rows row0..row0+15.
__device__ __forceinline__ v16bf load_a_frag(const __bf16* tile, int row0, int lane) {
    const int r = lane & 15, h = (lane >> 4) & 1;
    const __bf16* p = tile + (row0 + r) * LDK + h * 16;
    union { v4u u[2]; v16bf v; } f;
    f.u[0] = *(const v4u*)p;          // K [h*16,   h*16+8)
    f.u[1] = *(const v4u*)(p + 8);    // K [h*16+8, h*16+16)
    return f.v;
}

// B-matrix 32x16 bf16 fragment: lanes 0-15 = col N=lane, K 0..15 (V0-7);
// lanes 16-31 = col N=lane-16, K 16..31. LDS layout: tile[n*LDK + k].
__device__ __forceinline__ v16bf load_b_frag(const __bf16* tile, int col0, int lane) {
    const int n = lane & 15, h = (lane >> 4) & 1;
    const __bf16* p = tile + (col0 + n) * LDK + h * 16;
    union { v4u u[2]; v16bf v; } f;
    f.u[0] = *(const v4u*)p;          // K [h*16,   h*16+8)
    f.u[1] = *(const v4u*)(p + 8);    // K [h*16+8, h*16+16)
    return f.v;
}

// ---------------------------------------------------------------------------
// K0: zero the accumulators (colsumU, colsumV, VtZ, Dinv)
// ---------------------------------------------------------------------------
__global__ void k_init(float* __restrict__ p, int n) {
    int i = blockIdx.x * 256 + threadIdx.x;
    if (i < n) p[i] = 0.0f;
}

// ---------------------------------------------------------------------------
// K1: tmp = relu(x @ W^T + b); route columns to U/V/Z (ws) or T (out right
//     half); accumulate colsum(U), colsum(V).
// Block: 256 thr = 8 waves; tile BM=128 x BN=128; K-chunk 32.
// Wave tile 32(m) x 64(n) = 8 WMMA tiles -> ~1 wmma per ds_load_b128.
// Block columns lie in exactly one 256-col segment (128 | 256).
// ---------------------------------------------------------------------------
__global__ __launch_bounds__(256) void k_gemm_relu(
    const float* __restrict__ x, const float* __restrict__ W,
    const float* __restrict__ b,
    float* __restrict__ U, float* __restrict__ V, float* __restrict__ Z,
    float* __restrict__ out, float* __restrict__ colsumU,
    float* __restrict__ colsumV)
{
    __shared__ __align__(16) __bf16 sXh[128 * LDK], sXl[128 * LDK];
    __shared__ __align__(16) __bf16 sWh[128 * LDK], sWl[128 * LDK];
    __shared__ float sCol[128];
    __shared__ float sBias[128];

    const int tid  = threadIdx.x;
    const int lane = tid & 31;
    const int wave = tid >> 5;
    const int m0   = blockIdx.y * 128;
    const int nb   = blockIdx.x * 128;
    const int seg  = nb >> 8;          // 0:U 1:V 2:Z 3:T (uniform per block)
    const int segc = nb & 255;

    if (tid < 128) { sBias[tid] = b[nb + tid]; sCol[tid] = 0.0f; }

    const int wm = (wave >> 1) * 32;   // wave row offset in tile
    const int wn = (wave & 1) * 64;    // wave col offset in tile

    v8f acc[2][4] = {};

    for (int k0 = 0; k0 < DDIM; k0 += 32) {
        __syncthreads();
        // ---- fill phase: batch all global loads, then convert ----
        const int fr = tid >> 3;            // 0..31 (+32 per it)
        const int fc = (tid & 7) * 4;       // 0..28
        v4f xv[4], wv[4];
#pragma unroll
        for (int it = 0; it < 4; ++it) {
            const int r = fr + it * 32;
            xv[it] = *(const v4f*)(x + (size_t)(m0 + r) * DDIM + k0 + fc);
            wv[it] = *(const v4f*)(W + (size_t)(nb + r) * DDIM + k0 + fc);
        }
        if (k0 + 32 < DDIM) {               // prefetch next k-chunk (global_prefetch_b8)
#pragma unroll
            for (int it = 0; it < 4; ++it) {
                const int r = fr + it * 32;
                __builtin_prefetch(x + (size_t)(m0 + r) * DDIM + k0 + 32 + fc, 0, 0);
                __builtin_prefetch(W + (size_t)(nb + r) * DDIM + k0 + 32 + fc, 0, 0);
            }
        }
#pragma unroll
        for (int it = 0; it < 4; ++it) {
            const int r = fr + it * 32;
            __bf16 h, l;
            split_bf16(xv[it].x, h, l); sXh[r * LDK + fc + 0] = h; sXl[r * LDK + fc + 0] = l;
            split_bf16(xv[it].y, h, l); sXh[r * LDK + fc + 1] = h; sXl[r * LDK + fc + 1] = l;
            split_bf16(xv[it].z, h, l); sXh[r * LDK + fc + 2] = h; sXl[r * LDK + fc + 2] = l;
            split_bf16(xv[it].w, h, l); sXh[r * LDK + fc + 3] = h; sXl[r * LDK + fc + 3] = l;
            split_bf16(wv[it].x, h, l); sWh[r * LDK + fc + 0] = h; sWl[r * LDK + fc + 0] = l;
            split_bf16(wv[it].y, h, l); sWh[r * LDK + fc + 1] = h; sWl[r * LDK + fc + 1] = l;
            split_bf16(wv[it].z, h, l); sWh[r * LDK + fc + 2] = h; sWl[r * LDK + fc + 2] = l;
            split_bf16(wv[it].w, h, l); sWh[r * LDK + fc + 3] = h; sWl[r * LDK + fc + 3] = l;
        }
        __syncthreads();

        // ---- compute phase ----
        v16bf ah[2], al[2];
        ah[0] = load_a_frag(sXh, wm,      lane);
        ah[1] = load_a_frag(sXh, wm + 16, lane);
        al[0] = load_a_frag(sXl, wm,      lane);
        al[1] = load_a_frag(sXl, wm + 16, lane);
#pragma unroll
        for (int ni = 0; ni < 4; ++ni) {
            const v16bf bh = load_b_frag(sWh, wn + ni * 16, lane);
            const v16bf bl = load_b_frag(sWl, wn + ni * 16, lane);
#pragma unroll
            for (int mi = 0; mi < 2; ++mi) {
                acc[mi][ni] = wmma_bf16(ah[mi], bh, acc[mi][ni]);
                acc[mi][ni] = wmma_bf16(al[mi], bh, acc[mi][ni]);
                acc[mi][ni] = wmma_bf16(ah[mi], bl, acc[mi][ni]);
            }
        }
    }

    // Epilogue: bias + relu, route, column partial sums
    float lsum[4] = {0.0f, 0.0f, 0.0f, 0.0f};
#pragma unroll
    for (int mi = 0; mi < 2; ++mi)
#pragma unroll
        for (int ni = 0; ni < 4; ++ni) {
            const int lc = wn + ni * 16 + (lane & 15);     // col within block
            const float bb = sBias[lc];
#pragma unroll
            for (int r = 0; r < 8; ++r) {
                const int rowL = wm + mi * 16 + r + ((lane >> 4) ? 8 : 0);
                float v = acc[mi][ni][r] + bb;
                v = v > 0.0f ? v : 0.0f;
                const size_t grow = (size_t)(m0 + rowL);
                const int gc = segc + lc;                  // col within segment
                if (seg == 3) {
                    out[grow * 512 + 256 + gc] = v;        // T straight to output
                } else {
                    float* dst = (seg == 0) ? U : (seg == 1 ? V : Z);
                    dst[grow * KDIM + gc] = v;
                    if (seg < 2) lsum[ni] += v;
                }
            }
        }
    if (seg < 2) {
#pragma unroll
        for (int ni = 0; ni < 4; ++ni)
            atomicAdd(&sCol[wn + ni * 16 + (lane & 15)], lsum[ni]);  // ds_add_f32
        __syncthreads();
        if (tid < 128) {
            float* cs = (seg == 0) ? colsumU : colsumV;
            atomicAdd(&cs[segc + tid], sCol[tid]);         // global_atomic_add_f32
        }
    }
}

// ---------------------------------------------------------------------------
// K2: VtZ += V^T @ Z (split-K with global f32 atomics).
// grid (4 a-blocks, 64 row-chunks); BM=64(a) x BN=256(b); each block reduces
// 1024 rows of V/Z in 32-row steps. Wave: 16(a) x 128(b) = 8 WMMA tiles.
// ---------------------------------------------------------------------------
__global__ __launch_bounds__(256) void k_vtz(
    const float* __restrict__ V, const float* __restrict__ Z,
    float* __restrict__ VtZ)
{
    __shared__ __align__(16) __bf16 sAh[64 * LDK],  sAl[64 * LDK];
    __shared__ __align__(16) __bf16 sBh[256 * LDK], sBl[256 * LDK];

    const int tid  = threadIdx.x;
    const int lane = tid & 31;
    const int wave = tid >> 5;
    const int a0   = blockIdx.x * 64;
    const int i0b  = blockIdx.y * 1024;
    const int wa   = (wave & 3) * 16;    // a sub-row
    const int wb   = (wave >> 2) * 128;  // b block

    v8f acc[8] = {};

    for (int it = 0; it < 32; ++it) {
        const int i0 = i0b + it * 32;
        __syncthreads();
        // ---- fill phase: batch loads ----
        // A tile: 32(i) x 64(a) of V, stored transposed sA[a][i]
        const int ai = tid >> 4;            // 0..15 (+16 per iter)
        const int aa = (tid & 15) * 4;      // 0..60
        v4f av[2];
#pragma unroll
        for (int t = 0; t < 2; ++t)
            av[t] = *(const v4f*)(V + (size_t)(i0 + ai + t * 16) * KDIM + a0 + aa);
        // B tile: 32(i) x 256(b) of Z, stored transposed sB[b][i]
        const int bi = tid >> 6;            // 0..3 (+4 per iter)
        const int bb = (tid & 63) * 4;      // 0..252
        v4f bv[8];
#pragma unroll
        for (int t = 0; t < 8; ++t)
            bv[t] = *(const v4f*)(Z + (size_t)(i0 + bi + t * 4) * KDIM + bb);
#pragma unroll
        for (int t = 0; t < 2; ++t) {
            const int i = ai + t * 16;
            __bf16 h, l;
            split_bf16(av[t].x, h, l); sAh[(aa + 0) * LDK + i] = h; sAl[(aa + 0) * LDK + i] = l;
            split_bf16(av[t].y, h, l); sAh[(aa + 1) * LDK + i] = h; sAl[(aa + 1) * LDK + i] = l;
            split_bf16(av[t].z, h, l); sAh[(aa + 2) * LDK + i] = h; sAl[(aa + 2) * LDK + i] = l;
            split_bf16(av[t].w, h, l); sAh[(aa + 3) * LDK + i] = h; sAl[(aa + 3) * LDK + i] = l;
        }
#pragma unroll
        for (int t = 0; t < 8; ++t) {
            const int i = bi + t * 4;
            __bf16 h, l;
            split_bf16(bv[t].x, h, l); sBh[(bb + 0) * LDK + i] = h; sBl[(bb + 0) * LDK + i] = l;
            split_bf16(bv[t].y, h, l); sBh[(bb + 1) * LDK + i] = h; sBl[(bb + 1) * LDK + i] = l;
            split_bf16(bv[t].z, h, l); sBh[(bb + 2) * LDK + i] = h; sBl[(bb + 2) * LDK + i] = l;
            split_bf16(bv[t].w, h, l); sBh[(bb + 3) * LDK + i] = h; sBl[(bb + 3) * LDK + i] = l;
        }
        __syncthreads();

        const v16bf ah = load_a_frag(sAh, wa, lane);
        const v16bf al = load_a_frag(sAl, wa, lane);
#pragma unroll
        for (int nt = 0; nt < 8; ++nt) {
            const v16bf bh = load_b_frag(sBh, wb + nt * 16, lane);
            const v16bf bl = load_b_frag(sBl, wb + nt * 16, lane);
            acc[nt] = wmma_bf16(ah, bh, acc[nt]);
            acc[nt] = wmma_bf16(al, bh, acc[nt]);
            acc[nt] = wmma_bf16(ah, bl, acc[nt]);
        }
    }

#pragma unroll
    for (int nt = 0; nt < 8; ++nt)
#pragma unroll
        for (int r = 0; r < 8; ++r) {
            const int ar = a0 + wa + r + ((lane >> 4) ? 8 : 0);
            const int bc = wb + nt * 16 + (lane & 15);
            atomicAdd(&VtZ[(size_t)ar * KDIM + bc], acc[nt][r]);
        }
}

// ---------------------------------------------------------------------------
// K3: D = 1 / (colsum(U) . colsum(V) / n + eps)
// ---------------------------------------------------------------------------
__global__ void k_norm(const float* __restrict__ colsumU,
                       const float* __restrict__ colsumV,
                       float* __restrict__ Dinv)
{
    __shared__ float red[256];
    const int t = threadIdx.x;
    red[t] = colsumU[t] * colsumV[t];
    __syncthreads();
    for (int s = 128; s > 0; s >>= 1) {
        if (t < s) red[t] += red[t + s];
        __syncthreads();
    }
    if (t == 0) Dinv[0] = 1.0f / (red[0] / (float)NROWS + 1e-6f);
}

// ---------------------------------------------------------------------------
// K4: res = (U @ VtZ) * D -> left half of out.
// grid (2 n-blocks, 512 m-blocks); BM=128 x BN=128; K=256 in 32-chunks.
// Wave: 32(m) x 64(n) = 8 WMMA tiles.
// ---------------------------------------------------------------------------
__global__ __launch_bounds__(256) void k_res(
    const float* __restrict__ U, const float* __restrict__ VtZ,
    const float* __restrict__ Dinv, float* __restrict__ out)
{
    __shared__ __align__(16) __bf16 sAh[128 * LDK], sAl[128 * LDK];
    __shared__ __align__(16) __bf16 sBh[128 * LDK], sBl[128 * LDK];

    const int tid  = threadIdx.x;
    const int lane = tid & 31;
    const int wave = tid >> 5;
    const int m0   = blockIdx.y * 128;
    const int nb   = blockIdx.x * 128;
    const int wm   = (wave >> 1) * 32;
    const int wn   = (wave & 1) * 64;

    v8f acc[2][4] = {};

    for (int k0 = 0; k0 < KDIM; k0 += 32) {
        __syncthreads();
        // ---- fill phase: batch loads ----
        // A tile: U 128(m) x 32(k)
        const int fr = tid >> 3;            // 0..31 (+32 per it)
        const int fc = (tid & 7) * 4;       // 0..28
        v4f uv[4];
#pragma unroll
        for (int t = 0; t < 4; ++t)
            uv[t] = *(const v4f*)(U + (size_t)(m0 + fr + t * 32) * KDIM + k0 + fc);
        // B tile: VtZ 32(k) x 128(n), stored transposed sB[n][k]
        const int bk = tid >> 5;            // 0..7 (+8 per it)
        const int bn = (tid & 31) * 4;      // 0..124
        v4f bv[4];
#pragma unroll
        for (int t = 0; t < 4; ++t)
            bv[t] = *(const v4f*)(VtZ + (size_t)(k0 + bk + t * 8) * KDIM + nb + bn);
#pragma unroll
        for (int t = 0; t < 4; ++t) {
            const int r = fr + t * 32;
            __bf16 h, l;
            split_bf16(uv[t].x, h, l); sAh[r * LDK + fc + 0] = h; sAl[r * LDK + fc + 0] = l;
            split_bf16(uv[t].y, h, l); sAh[r * LDK + fc + 1] = h; sAl[r * LDK + fc + 1] = l;
            split_bf16(uv[t].z, h, l); sAh[r * LDK + fc + 2] = h; sAl[r * LDK + fc + 2] = l;
            split_bf16(uv[t].w, h, l); sAh[r * LDK + fc + 3] = h; sAl[r * LDK + fc + 3] = l;
        }
#pragma unroll
        for (int t = 0; t < 4; ++t) {
            const int k = bk + t * 8;
            __bf16 h, l;
            split_bf16(bv[t].x, h, l); sBh[(bn + 0) * LDK + k] = h; sBl[(bn + 0) * LDK + k] = l;
            split_bf16(bv[t].y, h, l); sBh[(bn + 1) * LDK + k] = h; sBl[(bn + 1) * LDK + k] = l;
            split_bf16(bv[t].z, h, l); sBh[(bn + 2) * LDK + k] = h; sBl[(bn + 2) * LDK + k] = l;
            split_bf16(bv[t].w, h, l); sBh[(bn + 3) * LDK + k] = h; sBl[(bn + 3) * LDK + k] = l;
        }
        __syncthreads();

        v16bf ah[2], al[2];
        ah[0] = load_a_frag(sAh, wm,      lane);
        ah[1] = load_a_frag(sAh, wm + 16, lane);
        al[0] = load_a_frag(sAl, wm,      lane);
        al[1] = load_a_frag(sAl, wm + 16, lane);
#pragma unroll
        for (int ni = 0; ni < 4; ++ni) {
            const v16bf bh = load_b_frag(sBh, wn + ni * 16, lane);
            const v16bf bl = load_b_frag(sBl, wn + ni * 16, lane);
#pragma unroll
            for (int mi = 0; mi < 2; ++mi) {
                acc[mi][ni] = wmma_bf16(ah[mi], bh, acc[mi][ni]);
                acc[mi][ni] = wmma_bf16(al[mi], bh, acc[mi][ni]);
                acc[mi][ni] = wmma_bf16(ah[mi], bl, acc[mi][ni]);
            }
        }
    }

    const float D = Dinv[0];
#pragma unroll
    for (int mi = 0; mi < 2; ++mi)
#pragma unroll
        for (int ni = 0; ni < 4; ++ni)
#pragma unroll
            for (int r = 0; r < 8; ++r) {
                const int row = m0 + wm + mi * 16 + r + ((lane >> 4) ? 8 : 0);
                const int col = nb + wn + ni * 16 + (lane & 15);
                out[(size_t)row * 512 + col] = acc[mi][ni][r] * D;
            }
}

// ---------------------------------------------------------------------------
extern "C" void kernel_launch(void* const* d_in, const int* in_sizes, int n_in,
                              void* d_out, int out_size, void* d_ws, size_t ws_size,
                              hipStream_t stream) {
    const float* x = (const float*)d_in[0];
    const float* W = (const float*)d_in[1];
    const float* b = (const float*)d_in[2];
    float* out = (float*)d_out;
    float* ws  = (float*)d_ws;

    const size_t NK = (size_t)NROWS * KDIM;       // 16M floats per matrix
    float* U       = ws;
    float* V       = ws + NK;
    float* Z       = ws + 2 * NK;
    float* colsumU = ws + 3 * NK;
    float* colsumV = colsumU + 256;
    float* VtZ     = colsumV + 256;
    float* Dinv    = VtZ + KDIM * KDIM;
    const int nAcc = 256 + 256 + KDIM * KDIM + 1; // 66049 floats to zero

    k_init<<<dim3((nAcc + 255) / 256), 256, 0, stream>>>(colsumU, nAcc);
    k_gemm_relu<<<dim3(DDIM / 128, NROWS / 128), 256, 0, stream>>>(
        x, W, b, U, V, Z, out, colsumU, colsumV);
    k_vtz<<<dim3(KDIM / 64, NROWS / 1024), 256, 0, stream>>>(V, Z, VtZ);
    k_norm<<<1, 256, 0, stream>>>(colsumU, colsumV, Dinv);
    k_res<<<dim3(KDIM / 128, NROWS / 128), 256, 0, stream>>>(U, VtZ, Dinv, out);
}
